// MultiheadAttention_7954279432591
// MI455X (gfx1250) — matrix operand order repaired and checked by reading, not compiled, not run on hardware
//
#include <hip/hip_runtime.h>
#include <stdint.h>

// B=8, S=1024, D=1024, H=16, DH=64
#define SDIM 1024
#define DDIM 1024
#define BNUM 8
#define DH   64

typedef __attribute__((ext_vector_type(16))) __bf16 v16bf;
typedef __attribute__((ext_vector_type(8)))  float  v8f;

union Frag { v16bf v; unsigned int u[8]; };

__device__ __forceinline__ unsigned short f2bf(float x) {
  unsigned int u = __float_as_uint(x);
  u += 0x7FFFu + ((u >> 16) & 1u);          // round-to-nearest-even
  return (unsigned short)(u >> 16);
}
__device__ __forceinline__ unsigned int packbf(float a, float b) {
  return (unsigned int)f2bf(a) | ((unsigned int)f2bf(b) << 16);
}

// LDS tile strides in bf16 units (padded to dodge bank conflicts)
#define AST 34
#define BST 34

// A-fragment: 16x32 bf16, lane holds row m=lane&15; VGPR p holds K pair
// k0 = (p>=4?16:0) + (lane>>4)*8 + (p&3)*2   (ISA 7.12.2 16-bit A layout)
__device__ __forceinline__ v16bf frag_a(const unsigned short* As, int m, int half) {
  Frag f;
  const unsigned short* row = As + m * AST;
#pragma unroll
  for (int p = 0; p < 8; ++p) {
    int k0 = ((p & 4) << 2) + (half << 3) + ((p & 3) << 1);
    f.u[p] = *(const unsigned int*)(row + k0);
  }
  return f.v;
}
// B-fragment from transposed LDS tile Bt[n][k]: lane holds col n=lane&15;
// VGPR p holds K pair k0 = (p>=4?16:0) + (lane>>4)*8 + (p&3)*2
__device__ __forceinline__ v16bf frag_b(const unsigned short* Bs, int n, int khalf) {
  Frag f;
  const unsigned short* row = Bs + n * BST;
#pragma unroll
  for (int p = 0; p < 8; ++p) {
    int k0 = ((p & 4) << 2) + (khalf << 3) + ((p & 3) << 1);
    f.u[p] = *(const unsigned int*)(row + k0);
  }
  return f.v;
}

// Stage 256x32 A tile (f32 -> packed bf16 pairs), one row per thread.
__device__ __forceinline__ void stageA(unsigned short* dstA, const float* __restrict__ A,
                                       int lda, int m0, int kc, int tid) {
  const float* src = A + (long long)(m0 + tid) * lda + kc;
  unsigned int* dst = (unsigned int*)(dstA + tid * AST);
#pragma unroll
  for (int q = 0; q < 8; ++q) {
    float4 x = ((const float4*)src)[q];
    dst[2 * q]     = packbf(x.x, x.y);
    dst[2 * q + 1] = packbf(x.z, x.w);
  }
}
// Stage 32(K)x64(N) B tile transposed -> Bt[n][k] (row-major [K,N] source).
__device__ __forceinline__ void stageB_nn(unsigned short* dstB, const float* __restrict__ Bm,
                                          int ldb, int n0, int kc, int tid) {
  int kk = tid >> 3, ns = (tid & 7) << 3;
  const float* src = Bm + (long long)(kc + kk) * ldb + n0 + ns;
  float4 x0 = ((const float4*)src)[0];
  float4 x1 = ((const float4*)src)[1];
  unsigned short vv[8] = { f2bf(x0.x), f2bf(x0.y), f2bf(x0.z), f2bf(x0.w),
                           f2bf(x1.x), f2bf(x1.y), f2bf(x1.z), f2bf(x1.w) };
#pragma unroll
  for (int x = 0; x < 8; ++x) dstB[(ns + x) * BST + kk] = vv[x];
}
// Stage Bt[n][k] from row-major [N,K] source (NT gemm: K^T is free).
__device__ __forceinline__ void stageB_nt(unsigned short* dstB, const float* __restrict__ Bm,
                                          int ldb, int n0, int kc, int tid) {
  int nn = tid >> 2, ks = (tid & 3) << 3;
  const float* src = Bm + (long long)(n0 + nn) * ldb + kc + ks;
  float4 y0 = ((const float4*)src)[0];
  float4 y1 = ((const float4*)src)[1];
  unsigned int* dst = (unsigned int*)(dstB + nn * BST + ks);
  dst[0] = packbf(y0.x, y0.y); dst[1] = packbf(y0.z, y0.w);
  dst[2] = packbf(y1.x, y1.y); dst[3] = packbf(y1.z, y1.w);
}

// Inner MMA step: preload 2 A-frags + all 4 B-frags into disjoint registers,
// then issue 8 back-to-back WMMAs (no intra-clause register hazards -> no
// compiler-inserted v_nops between matrix ops).
__device__ __forceinline__ void mma_step(const unsigned short* As, const unsigned short* Bs,
                                         int wbase, int lane, v8f acc[8]) {
  const int half = lane >> 4;
  const int l15  = lane & 15;
  v16bf a0 = frag_a(As, wbase + l15, half);
  v16bf a1 = frag_a(As, wbase + 16 + l15, half);
  v16bf b0 = frag_b(Bs, l15, half);
  v16bf b1 = frag_b(Bs, 16 + l15, half);
  v16bf b2 = frag_b(Bs, 32 + l15, half);
  v16bf b3 = frag_b(Bs, 48 + l15, half);
  acc[0] = __builtin_amdgcn_wmma_f32_16x16x32_bf16(false, a0, false, b0, (short)0, acc[0], false, false);
  acc[1] = __builtin_amdgcn_wmma_f32_16x16x32_bf16(false, a1, false, b0, (short)0, acc[1], false, false);
  acc[2] = __builtin_amdgcn_wmma_f32_16x16x32_bf16(false, a0, false, b1, (short)0, acc[2], false, false);
  acc[3] = __builtin_amdgcn_wmma_f32_16x16x32_bf16(false, a1, false, b1, (short)0, acc[3], false, false);
  acc[4] = __builtin_amdgcn_wmma_f32_16x16x32_bf16(false, a0, false, b2, (short)0, acc[4], false, false);
  acc[5] = __builtin_amdgcn_wmma_f32_16x16x32_bf16(false, a1, false, b2, (short)0, acc[5], false, false);
  acc[6] = __builtin_amdgcn_wmma_f32_16x16x32_bf16(false, a0, false, b3, (short)0, acc[6], false, false);
  acc[7] = __builtin_amdgcn_wmma_f32_16x16x32_bf16(false, a1, false, b3, (short)0, acc[7], false, false);
}

// ---------------------------------------------------------------------------
// Generic NN GEMM: C[z] = A[z][M,K] x Bm[z][K,N] (+bias) (+res), bf16 WMMA.
// Block tile 256(M) x 64(N), 8 waves; each wave 32x64 = 8 accumulators.
// Double-buffered LDS, Kc=32, prefetch of chunk kc+64 (global_prefetch_b8).
// ---------------------------------------------------------------------------
template<bool BIAS, bool RES>
__global__ __launch_bounds__(256)
void gemm_nn_kernel(const float* __restrict__ A, const float* __restrict__ Bm,
                    const float* __restrict__ bias, const float* __restrict__ res,
                    float* __restrict__ C, int Ktot, int lda, int ldb, int ldc,
                    long long sA, long long sB, long long sC) {
  __shared__ unsigned short As[2][256 * AST];
  __shared__ unsigned short Bs[2][64 * BST];
  const int tid   = threadIdx.x;
  const int lane  = tid & 31;
  const int wbase = (tid >> 5) << 5;       // wave's 32-row M base inside tile
  const int m0    = blockIdx.y * 256;
  const int n0    = blockIdx.x * 64;
  A  += (long long)blockIdx.z * sA;
  Bm += (long long)blockIdx.z * sB;
  C  += (long long)blockIdx.z * sC;

  v8f acc[8];
#pragma unroll
  for (int f = 0; f < 8; ++f)
#pragma unroll
    for (int r = 0; r < 8; ++r) acc[f][r] = 0.0f;

  stageA(As[0], A, lda, m0, 0, tid);
  stageB_nn(Bs[0], Bm, ldb, n0, 0, tid);
  __syncthreads();

  int cur = 0;
  for (int kc = 0; kc < Ktot; kc += 32) {
    if (kc + 64 < Ktot) {  // prefetch chunk-after-next (gfx1250 global_prefetch_b8)
      __builtin_prefetch(A + (long long)(m0 + tid) * lda + kc + 64, 0, 3);
      __builtin_prefetch(Bm + (long long)(kc + 64 + (tid >> 3)) * ldb + n0 + ((tid & 7) << 3), 0, 3);
    }
    int nxt = cur ^ 1;
    if (kc + 32 < Ktot) {  // stage next chunk into the other buffer
      stageA(As[nxt], A, lda, m0, kc + 32, tid);
      stageB_nn(Bs[nxt], Bm, ldb, n0, kc + 32, tid);
    }
    mma_step(As[cur], Bs[cur], wbase, lane, acc);
    __syncthreads();
    cur = nxt;
  }

  // epilogue: D layout per frag — m = r + (lane>>4)*8, n = lane&15
  // acc index: f-th B column pair -> acc[2f+am] covers rows am*16..am*16+15
#pragma unroll
  for (int f = 0; f < 4; ++f) {
    const int ng = n0 + (f << 4) + (lane & 15);
    const float bv = BIAS ? bias[ng] : 0.0f;
#pragma unroll
    for (int am = 0; am < 2; ++am)
#pragma unroll
      for (int r = 0; r < 8; ++r) {
        int mg = m0 + wbase + (am << 4) + r + ((lane >> 4) << 3);
        float val = acc[2 * f + am][r] + bv;
        if (RES) val += res[(long long)mg * ldc + ng];
        C[(long long)mg * ldc + ng] = val;
      }
  }
}

// ---------------------------------------------------------------------------
// Scores: per head-slice i (contiguous [1024,64]): S = 0.5 * Q_i x K_i^T,
// masked with atten_mask[i % 8] -> -1e6.  Same 256x64 tile machinery, K=64.
// ---------------------------------------------------------------------------
__global__ __launch_bounds__(256)
void scores_kernel(const float* __restrict__ Q, const float* __restrict__ Kmat,
                   const unsigned char* __restrict__ mask, float* __restrict__ Sc) {
  __shared__ unsigned short As[2][256 * AST];
  __shared__ unsigned short Bs[2][64 * BST];
  const int tid = threadIdx.x, lane = tid & 31, wbase = (tid >> 5) << 5;
  const int i  = blockIdx.z;
  const int m0 = blockIdx.y * 256, n0 = blockIdx.x * 64;
  const float* Qi = Q    + (long long)i * (SDIM * DH);
  const float* Ki = Kmat + (long long)i * (SDIM * DH);
  float* Si       = Sc   + (long long)i * (SDIM * SDIM);
  const unsigned char* Mi = mask + (long long)(i & 7) * (SDIM * SDIM);

  v8f acc[8];
#pragma unroll
  for (int f = 0; f < 8; ++f)
#pragma unroll
    for (int r = 0; r < 8; ++r) acc[f][r] = 0.0f;

  stageA(As[0], Qi, DH, m0, 0, tid);
  stageB_nt(Bs[0], Ki, DH, n0, 0, tid);
  __syncthreads();

  int cur = 0;
  for (int kc = 0; kc < DH; kc += 32) {
    int nxt = cur ^ 1;
    if (kc + 32 < DH) {
      stageA(As[nxt], Qi, DH, m0, kc + 32, tid);
      stageB_nt(Bs[nxt], Ki, DH, n0, kc + 32, tid);
    }
    mma_step(As[cur], Bs[cur], wbase, lane, acc);
    __syncthreads();
    cur = nxt;
  }

#pragma unroll
  for (int f = 0; f < 4; ++f) {
    const int ng = n0 + (f << 4) + (lane & 15);
#pragma unroll
    for (int am = 0; am < 2; ++am)
#pragma unroll
      for (int r = 0; r < 8; ++r) {
        int mg = m0 + wbase + (am << 4) + r + ((lane >> 4) << 3);
        float val = acc[2 * f + am][r] * 0.5f;   // scale = (DH//H)**-0.5 = 0.5
        long long off = (long long)mg * SDIM + ng;
        Si[off] = Mi[off] ? -1.0e6f : val;
      }
  }
}

// ---------------------------------------------------------------------------
// Softmax over rows of the [128*1024, 1024] attention tensor (in place).
// One wave per row; 32 elements per lane; shfl_xor reductions (wave32).
// ---------------------------------------------------------------------------
__global__ __launch_bounds__(256)
void softmax_kernel(float* __restrict__ attn) {
  const int tid = threadIdx.x, lane = tid & 31;
  const long long row = (long long)blockIdx.x * 8 + (tid >> 5);
  float* p = attn + row * SDIM;
  float v[32];
  float mx = -3.4e38f;
#pragma unroll
  for (int t = 0; t < 32; ++t) { v[t] = p[lane + (t << 5)]; mx = fmaxf(mx, v[t]); }
#pragma unroll
  for (int off = 16; off > 0; off >>= 1) mx = fmaxf(mx, __shfl_xor(mx, off, 32));
  float s = 0.0f;
#pragma unroll
  for (int t = 0; t < 32; ++t) { v[t] = __expf(v[t] - mx); s += v[t]; }
#pragma unroll
  for (int off = 16; off > 0; off >>= 1) s += __shfl_xor(s, off, 32);
  float inv = 1.0f / s;
#pragma unroll
  for (int t = 0; t < 32; ++t) p[lane + (t << 5)] = v[t] * inv;
}

// ---------------------------------------------------------------------------
// LayerNorm over last dim (1024): one wave per row.
// ---------------------------------------------------------------------------
__global__ __launch_bounds__(256)
void layernorm_kernel(const float* __restrict__ x, const float* __restrict__ gamma,
                      const float* __restrict__ beta, float* __restrict__ out) {
  const int tid = threadIdx.x, lane = tid & 31;
  const long long row = (long long)blockIdx.x * 8 + (tid >> 5);
  const float* p = x + row * DDIM;
  float v[32];
  float s = 0.0f, sq = 0.0f;
#pragma unroll
  for (int t = 0; t < 32; ++t) { v[t] = p[lane + (t << 5)]; s += v[t]; sq += v[t] * v[t]; }
#pragma unroll
  for (int off = 16; off > 0; off >>= 1) {
    s  += __shfl_xor(s, off, 32);
    sq += __shfl_xor(sq, off, 32);
  }
  float mean = s * (1.0f / (float)DDIM);
  float var  = sq * (1.0f / (float)DDIM) - mean * mean;
  float rstd = rsqrtf(var + 1e-5f);
#pragma unroll
  for (int t = 0; t < 32; ++t) {
    int c = lane + (t << 5);
    out[row * DDIM + c] = (v[t] - mean) * rstd * gamma[c] + beta[c];
  }
}

// ---------------------------------------------------------------------------
extern "C" void kernel_launch(void* const* d_in, const int* in_sizes, int n_in,
                              void* d_out, int out_size, void* d_ws, size_t ws_size,
                              hipStream_t stream) {
  (void)in_sizes; (void)n_in; (void)out_size;
  const float* key   = (const float*)d_in[0];
  const float* value = (const float*)d_in[1];
  const float* query = (const float*)d_in[2];
  const unsigned char* mask = (const unsigned char*)d_in[3];  // numpy bool = 1 byte
  const float* Wk = (const float*)d_in[4];  const float* bk = (const float*)d_in[5];
  const float* Wv = (const float*)d_in[6];  const float* bv = (const float*)d_in[7];
  const float* Wq = (const float*)d_in[8];  const float* bq = (const float*)d_in[9];
  const float* Wf = (const float*)d_in[10]; const float* bfv = (const float*)d_in[11];
  const float* gamma = (const float*)d_in[12];
  const float* beta  = (const float*)d_in[13];

  const long long QKV = (long long)BNUM * SDIM * DDIM;   // 8388608 elements
  float* out  = (float*)d_out;                           // [8,1024,1024]
  float* attn = out + QKV;                               // [128,1024,1024]

  if (ws_size < (size_t)4 * (size_t)QKV * sizeof(float)) return;
  float* qb = (float*)d_ws;       // q projection; later reused as pre-LN out
  float* kb = qb + QKV;
  float* vb = kb + QKV;
  float* cb = vb + QKV;           // context

  dim3 blk(256);

  // Q/K/V projections: [8192,1024] x [1024,1024] + bias (M tiles = 32)
  gemm_nn_kernel<true, false><<<dim3(16, 32, 1), blk, 0, stream>>>(
      query, Wq, bq, nullptr, qb, 1024, 1024, 1024, 1024, 0, 0, 0);
  gemm_nn_kernel<true, false><<<dim3(16, 32, 1), blk, 0, stream>>>(
      key, Wk, bk, nullptr, kb, 1024, 1024, 1024, 1024, 0, 0, 0);
  gemm_nn_kernel<true, false><<<dim3(16, 32, 1), blk, 0, stream>>>(
      value, Wv, bv, nullptr, vb, 1024, 1024, 1024, 1024, 0, 0, 0);

  // Scores + mask + scale, written straight into the attention output region
  scores_kernel<<<dim3(16, 4, 128), blk, 0, stream>>>(qb, kb, mask, attn);

  // Row softmax in place (128*1024 rows, 8 rows per block)
  softmax_kernel<<<dim3((128 * 1024) / 8), blk, 0, stream>>>(attn);

  // Context: per slice i: [1024,1024] x [1024,64]
  gemm_nn_kernel<false, false><<<dim3(1, 4, 128), blk, 0, stream>>>(
      attn, vb, nullptr, nullptr, cb, 1024, 1024, 64, 64,
      (long long)SDIM * SDIM, (long long)SDIM * DH, (long long)SDIM * DH);

  // Final projection + bias + residual -> pre-LN (reuse qb)
  gemm_nn_kernel<true, true><<<dim3(16, 32, 1), blk, 0, stream>>>(
      cb, Wf, bfv, query, qb, 1024, 1024, 1024, 1024, 0, 0, 0);

  // LayerNorm -> out
  layernorm_kernel<<<dim3(8192 / 8), blk, 0, stream>>>(qb, gamma, beta, out);
}
